// GroupedQueryAttention_13520557047974
// MI455X (gfx1250) — compile-verified, matrix operand order checked by reading, not run on hardware
//
#include <hip/hip_runtime.h>
#include <hip/hip_bf16.h>

// ---- problem constants (reference: B=2,S=2048,D=1024,HQ=16,HKV=4,HD=64) ----
#define B_    2
#define S_    2048
#define D_    1024
#define HQ_   16
#define HKV_  4
#define HD_   64
#define DKV_  (HKV_ * HD_)     // 256
#define MTOK_ (B_ * S_)        // 4096

typedef __attribute__((ext_vector_type(16))) __bf16 v16bf;
typedef __attribute__((ext_vector_type(8)))  float  v8f;

#define WMMA_BF16(a, b, c) \
  __builtin_amdgcn_wmma_f32_16x16x32_bf16(false, (a), false, (b), (short)0, (c), false, false)

#if defined(__has_builtin)
#if __has_builtin(__builtin_amdgcn_global_load_async_to_lds_b128)
#define HAVE_ASYNC_LDS 1
#endif
#endif
#ifndef HAVE_ASYNC_LDS
#define HAVE_ASYNC_LDS 0
#endif

// Types matching the async-load builtin's parameters:
// (int __vector(4) addrspace(1)*, int __vector(4) addrspace(3)*, Imm, Imm)
typedef int v4i_vs __attribute__((vector_size(16)));
#if HAVE_ASYNC_LDS
typedef __attribute__((address_space(1))) v4i_vs as1_v4i;
typedef __attribute__((address_space(3))) v4i_vs as3_v4i;
#endif

union Frag { v16bf v; unsigned u[8]; };

// Load a 16x32 bf16 operand fragment from a row-major matrix (global or LDS).
// CDNA5 16-bit A/B layout: lane = row (A) / col (B), lanes 0-15 hold
// K in {0..7,16..23}, lanes 16-31 hold K in {8..15,24..31}; each VGPR packs
// two consecutive K values -> one dword load per VGPR.
static __device__ __forceinline__ Frag load_frag_rm(const __bf16* __restrict__ p,
                                                    int row0, int k0, int ld) {
  Frag f;
  const int l  = threadIdx.x & 31;
  const int hs = l >> 4;
  const int m  = l & 15;
  const __bf16* rp = p + (size_t)(row0 + m) * ld + (k0 + hs * 8);
#pragma unroll
  for (int i = 0; i < 8; ++i) {
    const int kk = (i & 3) * 2 + (i >> 2) * 16;
    f.u[i] = *(const unsigned*)(rp + kk);
  }
  return f;
}

// 16-byte global -> LDS copy: async (ASYNCcnt-tracked, no VGPR round trip) when
// the gfx1250 builtin exists; plain load+store fallback otherwise.
static __device__ __forceinline__ void copy16_g2l(const __bf16* g, __bf16* l) {
#if HAVE_ASYNC_LDS
  __builtin_amdgcn_global_load_async_to_lds_b128(
      (as1_v4i*)(unsigned long long)(const void*)g,
      (as3_v4i*)(unsigned)(size_t)(void*)l,
      0, 0);
#else
  *(uint4*)l = *(const uint4*)g;
#endif
}

static __device__ __forceinline__ void wait_copies() {
#if HAVE_ASYNC_LDS
#if __has_builtin(__builtin_amdgcn_s_wait_asynccnt)
  __builtin_amdgcn_s_wait_asynccnt(0);
#else
  asm volatile("s_wait_asynccnt 0" ::: "memory");
#endif
#endif
}

// ---------------- cast / layout kernels ----------------
__global__ void k_cast_bf16(const float* __restrict__ in, __bf16* __restrict__ out, int n) {
  int i = blockIdx.x * blockDim.x + threadIdx.x;
  if (i < n) out[i] = (__bf16)in[i];
}

// in: [K][N] f32 (x @ W layout)  ->  out: [N][K] bf16 (row-major B^T for WMMA)
__global__ void k_castT_bf16(const float* __restrict__ in, __bf16* __restrict__ out,
                             int K, int N) {
  int i = blockIdx.x * blockDim.x + threadIdx.x;
  if (i >= N * K) return;
  int n = i / K, k = i - n * K;
  out[i] = (__bf16)in[(size_t)k * N + n];
}

// vb: [B*S][HKV*HD] -> vtb: [B][HKV][HD][S]   (so PV B-operand loads contiguously)
__global__ void k_transpose_v(const __bf16* __restrict__ vb, __bf16* __restrict__ vtb) {
  int i = blockIdx.x * blockDim.x + threadIdx.x;
  if (i >= B_ * S_ * DKV_) return;
  int s    = i % S_;
  int rest = i / S_;                  // b*HKV*HD + h*HD + d
  int d    = rest % HD_;
  int h    = (rest / HD_) % HKV_;
  int b    = rest / (HD_ * HKV_);
  vtb[i] = vb[(size_t)(b * S_ + s) * DKV_ + h * HD_ + d];
}

// ---------------- generic WMMA GEMM: C[M,N] = A[M,K] x Bt[N,K]^T ----------------
// One wave computes a 16(M) x 64(N) tile; 4 f32 accumulators.
__global__ __launch_bounds__(128) void k_gemm_wmma(
    const __bf16* __restrict__ A, const __bf16* __restrict__ Bt,
    __bf16* __restrict__ outb, float* __restrict__ outf,
    int M, int N, int K) {
  const int wid     = (int)((blockIdx.x * blockDim.x + threadIdx.x) >> 5);
  const int tiles_n = N >> 6;
  if (wid >= (M >> 4) * tiles_n) return;
  const int row0 = (wid / tiles_n) << 4;
  const int n0   = (wid % tiles_n) << 6;
  const int l = threadIdx.x & 31, hs = l >> 4, nn = l & 15;

  v8f acc[4];
#pragma unroll
  for (int t = 0; t < 4; ++t) acc[t] = v8f{0.f, 0.f, 0.f, 0.f, 0.f, 0.f, 0.f, 0.f};

#pragma unroll 1
  for (int kk = 0; kk < K; kk += 32) {
    Frag a = load_frag_rm(A, row0, kk, K);
#pragma unroll
    for (int t = 0; t < 4; ++t) {
      Frag b = load_frag_rm(Bt, n0 + 16 * t, kk, K);
      acc[t] = WMMA_BF16(a.v, b.v, acc[t]);
    }
  }

  if (outb) {
#pragma unroll
    for (int t = 0; t < 4; ++t)
#pragma unroll
      for (int j = 0; j < 8; ++j)
        outb[(size_t)(row0 + j + hs * 8) * N + (n0 + 16 * t + nn)] = (__bf16)acc[t][j];
  } else {
#pragma unroll
    for (int t = 0; t < 4; ++t)
#pragma unroll
      for (int j = 0; j < 8; ++j)
        outf[(size_t)(row0 + j + hs * 8) * N + (n0 + 16 * t + nn)] = acc[t][j];
  }
}

// ---------------- flash attention (GQA, bias + mask, bf16 WMMA) ----------------
// One block = 4 waves sharing (b, hq): waves handle 4 consecutive 16-query
// tiles, and cooperatively stage each 32-key K block (32x64) and V^T block
// (64x32) into LDS via async global->LDS copies.
__global__ __launch_bounds__(128) void k_attention(
    const __bf16* __restrict__ qb,  const __bf16* __restrict__ kb,
    const __bf16* __restrict__ vtb, const float* __restrict__ pos_bias,
    const float* __restrict__ mask, __bf16* __restrict__ ab) {
  __shared__ __bf16 kbuf[32][64];      // 4 KB: K block, rows = keys
  __shared__ __bf16 vbuf[64][32];      // 4 KB: V^T block, rows = hd
  __shared__ __bf16 pbuf[4][16][32];   // 4 KB: per-wave P staging

  const int tid = threadIdx.x;
  const int w   = tid >> 5;
  const int qg  = blockIdx.x & 31;               // S_/(16*4) = 32 q-groups
  const int h   = (blockIdx.x >> 5) & 15;
  const int b   = blockIdx.x >> 9;
  const int hk  = h >> 2;                        // repeat_interleave: hq -> kv head
  const int q0  = ((qg << 2) + w) << 4;
  const int l = tid & 31, hs = l >> 4, nn = l & 15;

  const int rowA = b * S_ + q0;
  Frag aq0 = load_frag_rm(qb, rowA, h * HD_,      D_);
  Frag aq1 = load_frag_rm(qb, rowA, h * HD_ + 32, D_);

  float m_i[8], l_i[8];
#pragma unroll
  for (int j = 0; j < 8; ++j) { m_i[j] = -1e30f; l_i[j] = 0.f; }
  v8f acc[4];
#pragma unroll
  for (int t = 0; t < 4; ++t) acc[t] = v8f{0.f, 0.f, 0.f, 0.f, 0.f, 0.f, 0.f, 0.f};

  const __bf16* vt_base = vtb + (size_t)(b * HKV_ + hk) * HD_ * S_;

  // per-thread staging assignments (constant across the key loop)
  const int krow = tid >> 2, kseg = (tid & 3) * 16;   // K: 32 rows x 128 B
  const int vrow = tid >> 1, vseg = (tid & 1) * 16;   // V^T: 64 rows x 64 B
  const __bf16* ksrc0 = kb + (size_t)(b * S_ + krow) * DKV_ + hk * HD_ + kseg;
  const __bf16* vsrc0 = vt_base + (size_t)vrow * S_ + vseg;

#pragma unroll 1
  for (int k0 = 0; k0 < S_; k0 += 32) {
    __syncthreads();                       // previous block fully consumed
    // cooperative stage: K block + V^T block (32 B per thread per matrix)
    {
      const __bf16* ks = ksrc0 + (size_t)k0 * DKV_;
      copy16_g2l(ks,     &kbuf[krow][kseg]);
      copy16_g2l(ks + 8, &kbuf[krow][kseg + 8]);
      const __bf16* vs = vsrc0 + k0;
      copy16_g2l(vs,     &vbuf[vrow][vseg]);
      copy16_g2l(vs + 8, &vbuf[vrow][vseg + 8]);
    }
    // prefetch next block's bias row for this lane's first row
    if (k0 + 32 < S_)
      __builtin_prefetch(&pos_bias[((size_t)h * S_ + q0 + hs * 8) * S_ + k0 + 32], 0, 0);
    wait_copies();
    __syncthreads();                       // staged data visible to all waves

    float p[2][8];
#pragma unroll
    for (int t = 0; t < 2; ++t) {
      Frag bk0 = load_frag_rm(&kbuf[0][0], 16 * t, 0,  64);
      Frag bk1 = load_frag_rm(&kbuf[0][0], 16 * t, 32, 64);
      v8f z = v8f{0.f, 0.f, 0.f, 0.f, 0.f, 0.f, 0.f, 0.f};
      z = WMMA_BF16(aq0.v, bk0.v, z);
      z = WMMA_BF16(aq1.v, bk1.v, z);
      const int key = k0 + 16 * t + nn;
#pragma unroll
      for (int j = 0; j < 8; ++j) {
        const int row = q0 + j + hs * 8;
        p[t][j] = z[j] * 0.125f                                    // 1/sqrt(64)
                + pos_bias[((size_t)h * S_ + row) * S_ + key]
                + mask[(size_t)row * S_ + key];
      }
    }
    // online-softmax update (row stats replicated across the 16 lanes of a half)
#pragma unroll
    for (int j = 0; j < 8; ++j) {
      float bm = fmaxf(p[0][j], p[1][j]);
#pragma unroll
      for (int off = 8; off; off >>= 1) bm = fmaxf(bm, __shfl_xor(bm, off));
      const float mn    = fmaxf(m_i[j], bm);
      const float alpha = __expf(m_i[j] - mn);
      m_i[j] = mn;
      p[0][j] = __expf(p[0][j] - mn);
      p[1][j] = __expf(p[1][j] - mn);
      float rs = p[0][j] + p[1][j];
#pragma unroll
      for (int off = 8; off; off >>= 1) rs += __shfl_xor(rs, off);
      l_i[j] = l_i[j] * alpha + rs;
#pragma unroll
      for (int t2 = 0; t2 < 4; ++t2) acc[t2][j] *= alpha;
    }
    // C-layout -> A-fragment layout via per-wave LDS staging (DS ops in-order per wave)
#pragma unroll
    for (int t = 0; t < 2; ++t)
#pragma unroll
      for (int j = 0; j < 8; ++j)
        pbuf[w][j + hs * 8][16 * t + nn] = (__bf16)p[t][j];
    __asm volatile("" ::: "memory");
    Frag pa;
    {
      const __bf16* rp = &pbuf[w][nn][hs * 8];
#pragma unroll
      for (int i = 0; i < 8; ++i) {
        const int kk = (i & 3) * 2 + (i >> 2) * 16;
        pa.u[i] = *(const unsigned*)(rp + kk);
      }
    }
    __asm volatile("" ::: "memory");
    // acc += P[16x32] @ V[32x64]  (B operand from staged V^T block)
#pragma unroll
    for (int t2 = 0; t2 < 4; ++t2) {
      Frag bv = load_frag_rm(&vbuf[0][0], 16 * t2, 0, 32);
      acc[t2] = WMMA_BF16(pa.v, bv.v, acc[t2]);
    }
  }
  // epilogue: normalize and store bf16 [tok][D] for the output projection
#pragma unroll
  for (int j = 0; j < 8; ++j) {
    const float inv = 1.0f / l_i[j];
    const int   row = rowA + j + hs * 8;
#pragma unroll
    for (int t2 = 0; t2 < 4; ++t2)
      ab[(size_t)row * D_ + h * HD_ + 16 * t2 + nn] = (__bf16)(acc[t2][j] * inv);
  }
}

// ---------------- launcher ----------------
extern "C" void kernel_launch(void* const* d_in, const int* in_sizes, int n_in,
                              void* d_out, int out_size, void* d_ws, size_t ws_size,
                              hipStream_t stream) {
  const float* x  = (const float*)d_in[0];
  const float* wq = (const float*)d_in[1];
  const float* wk = (const float*)d_in[2];
  const float* wv = (const float*)d_in[3];
  const float* wo = (const float*)d_in[4];
  const float* pb = (const float*)d_in[5];
  const float* mk = (const float*)d_in[6];
  float* out = (float*)d_out;

  char* ws = (char*)d_ws;
  size_t off = 0;
  auto alloc = [&](size_t bytes) -> void* {
    void* p = ws + off;
    off += (bytes + 255) & ~(size_t)255;
    return p;
  };
  __bf16* xb  = (__bf16*)alloc((size_t)MTOK_ * D_   * 2);
  __bf16* wqT = (__bf16*)alloc((size_t)D_   * D_    * 2);
  __bf16* wkT = (__bf16*)alloc((size_t)D_   * DKV_  * 2);
  __bf16* wvT = (__bf16*)alloc((size_t)D_   * DKV_  * 2);
  __bf16* woT = (__bf16*)alloc((size_t)D_   * D_    * 2);
  __bf16* qb  = (__bf16*)alloc((size_t)MTOK_ * D_   * 2);
  __bf16* kb  = (__bf16*)alloc((size_t)MTOK_ * DKV_ * 2);
  __bf16* vb  = (__bf16*)alloc((size_t)MTOK_ * DKV_ * 2);
  __bf16* vtb = (__bf16*)alloc((size_t)MTOK_ * DKV_ * 2);
  __bf16* ab  = (__bf16*)alloc((size_t)MTOK_ * D_   * 2);
  (void)ws_size; (void)in_sizes; (void)n_in; (void)out_size;

  // 1) casts / weight transposes
  k_cast_bf16<<<(MTOK_ * D_ + 255) / 256, 256, 0, stream>>>(x, xb, MTOK_ * D_);
  k_castT_bf16<<<(D_ * D_   + 255) / 256, 256, 0, stream>>>(wq, wqT, D_, D_);
  k_castT_bf16<<<(D_ * DKV_ + 255) / 256, 256, 0, stream>>>(wk, wkT, D_, DKV_);
  k_castT_bf16<<<(D_ * DKV_ + 255) / 256, 256, 0, stream>>>(wv, wvT, D_, DKV_);
  k_castT_bf16<<<(D_ * D_   + 255) / 256, 256, 0, stream>>>(wo, woT, D_, D_);

  // 2) projections (wave tiles: (M/16)*(N/64); 4 waves per 128-thread block)
  {
    int tiles = (MTOK_ / 16) * (D_ / 64);        // 4096 waves
    k_gemm_wmma<<<(tiles + 3) / 4, 128, 0, stream>>>(xb, wqT, qb, nullptr, MTOK_, D_, D_);
  }
  {
    int tiles = (MTOK_ / 16) * (DKV_ / 64);      // 1024 waves
    k_gemm_wmma<<<(tiles + 3) / 4, 128, 0, stream>>>(xb, wkT, kb, nullptr, MTOK_, DKV_, D_);
    k_gemm_wmma<<<(tiles + 3) / 4, 128, 0, stream>>>(xb, wvT, vb, nullptr, MTOK_, DKV_, D_);
  }
  k_transpose_v<<<(MTOK_ * DKV_ + 255) / 256, 256, 0, stream>>>(vb, vtb);

  // 3) attention: one block per (b, hq, 64-query group) = 2*16*32 = 1024 blocks
  k_attention<<<B_ * HQ_ * (S_ / 64), 128, 0, stream>>>(qb, kb, vtb, pb, mk, ab);

  // 4) output projection -> f32 d_out
  {
    int tiles = (MTOK_ / 16) * (D_ / 64);
    k_gemm_wmma<<<(tiles + 3) / 4, 128, 0, stream>>>(ab, woT, nullptr, out, MTOK_, D_, D_);
  }
}
